// SS2D_8177617731787
// MI455X (gfx1250) — matrix operand order repaired and checked by reading, not compiled, not run on hardware
//
#include <hip/hip_runtime.h>

#define DEV __device__ __forceinline__

typedef _Float16 v16h __attribute__((ext_vector_type(16)));
typedef float    v8f  __attribute__((ext_vector_type(8)));

union HFrag { v16h v; unsigned u[8]; };

static constexpr int Bb = 4, L = 4096;
static constexpr int DM = 96, DSn = 16, DI = 192, RK = 6, K4 = 4;
static constexpr int CP = RK + 2 * DSn; // 38 proj rows

// workspace layout (floats)
static constexpr size_t SZ_BDL   = (size_t)Bb * DI * L;          // 3.1M
static constexpr size_t OFF_XIN  = 0;                            // [b][d][p]
static constexpr size_t OFF_Z    = OFF_XIN + SZ_BDL;             // [b*L+p][d]
static constexpr size_t OFF_XC   = OFF_Z + SZ_BDL;               // [b][d][p]
static constexpr size_t OFF_XDBL = OFF_XC + SZ_BDL;              // [b][k][c][p]
static constexpr size_t SZ_XDBL  = (size_t)Bb * K4 * CP * L;
static constexpr size_t OFF_DEL  = OFF_XDBL + SZ_XDBL;           // delta -> ys in place
static constexpr size_t SZ_DEL   = (size_t)Bb * K4 * DI * L;
static constexpr size_t OFF_YLN  = OFF_DEL + SZ_DEL;             // [b*L+p][d]

DEV unsigned pack2h(float a, float b) {
  union { _Float16 h[2]; unsigned u; } r;
  r.h[0] = (_Float16)a; r.h[1] = (_Float16)b;
  return r.u;
}

DEV v8f vzero() {
  v8f z;
#pragma unroll
  for (int i = 0; i < 8; ++i) z[i] = 0.f;
  return z;
}

DEV v8f wmma_f16(const HFrag& a, const HFrag& b, v8f c) {
  return __builtin_amdgcn_wmma_f32_16x16x32_f16(
      false, a.v, false, b.v, (short)0, c, false, false);
}

// A fragment (16x32 f16) from row-major LDS tile, stride in halves (even)
DEV void load_afrag(const _Float16* lds, int rowBase, int stride, int lane, HFrag& f) {
  int r = lane & 15, hf = lane >> 4;
  const _Float16* rp = lds + (rowBase + r) * stride;
#pragma unroll
  for (int v = 0; v < 8; ++v) {
    int k = 2 * v + 16 * hf;
    f.u[v] = *(const unsigned*)(rp + k);
  }
}

// B fragment (32x16 f16) from K-pair-packed LDS tile [16][ncols] uints
DEV void load_bfrag(const unsigned* ldsPk, int ncols, int colBase, int lane, HFrag& f) {
  int n = colBase + (lane & 15), hf = lane >> 4;
#pragma unroll
  for (int v = 0; v < 8; ++v) f.u[v] = ldsPk[(hf * 8 + v) * ncols + n];
}

// ---------------- Kernel 1: in_proj GEMM [16384x96]x[96x384], split xin/z
__global__ void k_inproj(const float* __restrict__ x, const float* __restrict__ w,
                         float* __restrict__ ws) {
  __shared__ _Float16 sA[64 * 34];
  __shared__ unsigned sB[16 * 128];
  float* XIN = ws + OFF_XIN;
  float* Z   = ws + OFF_Z;
  const int tid = threadIdx.x;
  const int m0 = blockIdx.x * 64, n0 = blockIdx.y * 128;
  const int wv = tid >> 5, lane = tid & 31;
  const int mt = wv >> 1, nb = (wv & 1) * 4;
  v8f acc[4];
  for (int i = 0; i < 4; ++i) acc[i] = vzero();
  for (int kc = 0; kc < 3; ++kc) {
    const int c0 = kc * 32;
#pragma unroll
    for (int i = 0; i < 8; ++i) {            // A 64x32
      int idx = tid + i * 256, ml = idx >> 5, c = idx & 31;
      sA[ml * 34 + c] = (_Float16)x[(size_t)(m0 + ml) * DM + c0 + c];
    }
#pragma unroll
    for (int i = 0; i < 8; ++i) {            // B 32x128 (K-pair packed)
      int idx = tid + i * 256, n = idx & 127, k2 = idx >> 7;
      const float* wp = w + (size_t)(n0 + n) * DM + c0 + k2 * 2;
      sB[k2 * 128 + n] = pack2h(wp[0], wp[1]);
    }
    __syncthreads();
    HFrag af; load_afrag(sA, mt * 16, 34, lane, af);
#pragma unroll
    for (int nt = 0; nt < 4; ++nt) {
      HFrag bf; load_bfrag(sB, 128, (nb + nt) * 16, lane, bf);
      acc[nt] = wmma_f16(af, bf, acc[nt]);
    }
    __syncthreads();
  }
  const int r = lane & 15, hf = lane >> 4;
#pragma unroll
  for (int nt = 0; nt < 4; ++nt) {
    int col = n0 + (nb + nt) * 16 + r;
#pragma unroll
    for (int v = 0; v < 8; ++v) {
      int m = m0 + mt * 16 + hf * 8 + v;
      int b = m >> 12, p = m & 4095;
      float val = acc[nt][v];
      if (col < DI) XIN[((size_t)b * DI + col) * L + p] = val;
      else          Z[(size_t)m * DI + (col - DI)] = val;
    }
  }
}

// ---------------- Kernel 2: depthwise 3x3 conv + bias + SiLU
__global__ void k_conv(const float* __restrict__ cw, const float* __restrict__ cb,
                       float* __restrict__ ws) {
  size_t idx = (size_t)blockIdx.x * 256 + threadIdx.x;
  const float* XIN = ws + OFF_XIN;
  float* XC = ws + OFF_XC;
  int p = (int)(idx & 4095); int rest = (int)(idx >> 12);
  int d = rest % DI, b = rest / DI;
  int h = p >> 6, wc = p & 63;
  const float* src = XIN + ((size_t)b * DI + d) * L;
  const float* wt = cw + d * 9;
  float s = cb[d];
#pragma unroll
  for (int kh = 0; kh < 3; ++kh) {
    int hh = h + kh - 1; if (hh < 0 || hh >= 64) continue;
#pragma unroll
    for (int kw = 0; kw < 3; ++kw) {
      int ww = wc + kw - 1; if (ww < 0 || ww >= 64) continue;
      s += src[hh * 64 + ww] * wt[kh * 3 + kw];
    }
  }
  XC[((size_t)b * DI + d) * L + p] = s / (1.f + __expf(-s));
}

// ---------------- Kernel 3a: x_proj GEMM per (b,k): [38x192]x[192x4096]
__global__ void k_xproj(const float* __restrict__ xw, float* __restrict__ ws) {
  __shared__ _Float16 sA[48 * 34];
  __shared__ unsigned sB[16 * 64];
  const float* XC = ws + OFF_XC;
  float* XDBL = ws + OFF_XDBL;
  const int tid = threadIdx.x;                 // 128 threads, 4 waves
  const int p0 = blockIdx.x * 64;
  const int k = blockIdx.y, b = blockIdx.z;
  const int wv = tid >> 5, lane = tid & 31;
  v8f acc[3];
  for (int i = 0; i < 3; ++i) acc[i] = vzero();
  for (int kc = 0; kc < 6; ++kc) {
    const int d0 = kc * 32;
#pragma unroll
    for (int i = 0; i < 12; ++i) {             // A 48x32, rows>=38 zero
      int idx = tid + i * 128, dl = idx & 31, c = idx >> 5;
      float v = (c < CP) ? xw[((size_t)k * CP + c) * DI + d0 + dl] : 0.f;
      sA[c * 34 + dl] = (_Float16)v;
    }
#pragma unroll
    for (int i = 0; i < 8; ++i) {              // B 32x64 packed
      int idx = tid + i * 128, n = idx & 63, k2 = idx >> 6;
      const float* xp = XC + ((size_t)b * DI + d0 + k2 * 2) * L + p0 + n;
      sB[k2 * 64 + n] = pack2h(xp[0], xp[L]);
    }
    __syncthreads();
    HFrag bf; load_bfrag(sB, 64, wv * 16, lane, bf);
#pragma unroll
    for (int ct = 0; ct < 3; ++ct) {
      HFrag af; load_afrag(sA, ct * 16, 34, lane, af);
      acc[ct] = wmma_f16(af, bf, acc[ct]);
    }
    __syncthreads();
  }
  const int r = lane & 15, hf = lane >> 4;
  const int col = p0 + wv * 16 + r;
#pragma unroll
  for (int ct = 0; ct < 3; ++ct)
#pragma unroll
    for (int v = 0; v < 8; ++v) {
      int c = ct * 16 + hf * 8 + v;
      if (c < CP)
        XDBL[((size_t)(b * K4 + k) * CP + c) * L + col] = acc[ct][v];
    }
}

// ---------------- Kernel 3b: dt_proj (K=6) + softplus -> delta
__global__ void k_delta(const float* __restrict__ dtw, const float* __restrict__ dtb,
                        float* __restrict__ ws) {
  size_t idx = (size_t)blockIdx.x * 256 + threadIdx.x;
  int p = (int)(idx & 4095); int rest = (int)(idx >> 12);
  int d = rest % DI; rest /= DI;
  int k = rest & 3, b = rest >> 2;
  const float* xr = ws + OFF_XDBL + ((size_t)(b * K4 + k) * CP) * L + p;
  const float* wr = dtw + ((size_t)k * DI + d) * RK;
  float s = dtb[k * DI + d];
#pragma unroll
  for (int r = 0; r < RK; ++r) s += xr[(size_t)r * L] * wr[r];
  float sp = (s > 20.f) ? s : log1pf(__expf(s));
  ws[OFF_DEL + ((size_t)(b * K4 + k) * DI + d) * L + p] = sp;
}

// ---------------- Kernel 4: selective scan, 16 states on 16 lanes, y scattered
// back to spatial order in place of delta.
__global__ void k_scan(const float* __restrict__ alog, float* __restrict__ ws) {
  const int dblk = blockIdx.x, k = blockIdx.y, b = blockIdx.z;
  const int tid = threadIdx.x;
  const int chain = tid >> 4, n = tid & 15;
  const int d = dblk * 16 + chain;
  const float An = -__expf(alog[((size_t)k * DI + d) * DSn + n]);
  float* drow = ws + OFF_DEL + ((size_t)(b * K4 + k) * DI + d) * L;
  const float* urow = ws + OFF_XC + ((size_t)b * DI + d) * L;
  const float* brow = ws + OFF_XDBL + ((size_t)(b * K4 + k) * CP + RK + n) * L;
  const float* crow = ws + OFF_XDBL + ((size_t)(b * K4 + k) * CP + RK + DSn + n) * L;
  const bool rev = (k >= 2), tr = (k & 1);
  float h = 0.f;
  for (int t = 0; t < L; ++t) {
    int tt = rev ? (L - 1 - t) : t;
    int p = tr ? (((tt & 63) << 6) | (tt >> 6)) : tt;   // transpose perm (64x64)
    if ((t & 15) == 0 && t + 32 < L) {                  // cover latency ahead
      int t2 = t + 32;
      int tt2 = rev ? (L - 1 - t2) : t2;
      int p2 = tr ? (((tt2 & 63) << 6) | (tt2 >> 6)) : tt2;
      __builtin_prefetch(brow + p2, 0, 0);
      __builtin_prefetch(crow + p2, 0, 0);
      __builtin_prefetch(drow + p2, 0, 0);
      __builtin_prefetch(urow + p2, 0, 0);
    }
    float dl = drow[p];
    float u  = urow[p];
    float bv = brow[p];
    float cv = crow[p];
    h = __expf(dl * An) * h + (dl * u) * bv;
    float y = h * cv;                                    // reduce over 16 states
    y += __shfl_xor(y, 8, 16);
    y += __shfl_xor(y, 4, 16);
    y += __shfl_xor(y, 2, 16);
    y += __shfl_xor(y, 1, 16);
    if (n == 0) drow[p] = y;                             // ys at spatial p
  }
}

// ---------------- Kernel 5: sum 4 dirs + D*u + LayerNorm + SiLU(z) gate
__global__ void k_combine(const float* __restrict__ Ds, const float* __restrict__ g,
                          const float* __restrict__ be, float* __restrict__ ws) {
  __shared__ float s1[DI], s2[DI];
  __shared__ float smu, srv;
  const int p = blockIdx.x, b = blockIdx.y, d = threadIdx.x;
  const float* YS = ws + OFF_DEL;
  const float* XC = ws + OFF_XC;
  const float* Z  = ws + OFF_Z;
  float y = 0.f;
#pragma unroll
  for (int k = 0; k < K4; ++k) y += YS[((size_t)(b * K4 + k) * DI + d) * L + p];
  float dsum = Ds[d] + Ds[DI + d] + Ds[2 * DI + d] + Ds[3 * DI + d];
  y += dsum * XC[((size_t)b * DI + d) * L + p];
  s1[d] = y; s2[d] = y * y;
  __syncthreads();
  for (int off = 96; off >= 3; off >>= 1) {
    if (d < off) { s1[d] += s1[d + off]; s2[d] += s2[d + off]; }
    __syncthreads();
  }
  if (d == 0) {
    float m = (s1[0] + s1[1] + s1[2]) * (1.f / DI);
    float v = (s2[0] + s2[1] + s2[2]) * (1.f / DI) - m * m;
    smu = m; srv = rsqrtf(v + 1e-5f);
  }
  __syncthreads();
  float yn = (y - smu) * srv * g[d] + be[d];
  float zv = Z[((size_t)b * L + p) * DI + d];
  yn *= zv / (1.f + __expf(-zv));
  ws[OFF_YLN + ((size_t)b * L + p) * DI + d] = yn;
}

// ---------------- Kernel 6: out_proj GEMM [16384x192]x[192x96]
__global__ void k_outproj(const float* __restrict__ wo, const float* __restrict__ wsp,
                          float* __restrict__ out) {
  __shared__ _Float16 sA[64 * 34];
  __shared__ unsigned sB[16 * 96];
  const float* Y = wsp + OFF_YLN;
  const int tid = threadIdx.x;
  const int m0 = blockIdx.x * 64;
  const int wv = tid >> 5, lane = tid & 31;
  const int mt = wv >> 1, nb = (wv & 1) * 3;
  v8f acc[3];
  for (int i = 0; i < 3; ++i) acc[i] = vzero();
  for (int kc = 0; kc < 6; ++kc) {
    const int c0 = kc * 32;
#pragma unroll
    for (int i = 0; i < 8; ++i) {            // A 64x32
      int idx = tid + i * 256, ml = idx >> 5, c = idx & 31;
      sA[ml * 34 + c] = (_Float16)Y[(size_t)(m0 + ml) * DI + c0 + c];
    }
#pragma unroll
    for (int i = 0; i < 6; ++i) {            // B 32x96 packed
      int idx = tid + i * 256, n = idx % 96, k2 = idx / 96;
      const float* wp = wo + (size_t)n * DI + c0 + k2 * 2;
      sB[k2 * 96 + n] = pack2h(wp[0], wp[1]);
    }
    __syncthreads();
    HFrag af; load_afrag(sA, mt * 16, 34, lane, af);
#pragma unroll
    for (int j = 0; j < 3; ++j) {
      HFrag bf; load_bfrag(sB, 96, (nb + j) * 16, lane, bf);
      acc[j] = wmma_f16(af, bf, acc[j]);
    }
    __syncthreads();
  }
  const int r = lane & 15, hf = lane >> 4;
#pragma unroll
  for (int j = 0; j < 3; ++j)
#pragma unroll
    for (int v = 0; v < 8; ++v) {
      int m = m0 + mt * 16 + hf * 8 + v;
      int c = (nb + j) * 16 + r;
      out[(size_t)m * DM + c] = acc[j][v];
    }
}

extern "C" void kernel_launch(void* const* d_in, const int* in_sizes, int n_in,
                              void* d_out, int out_size, void* d_ws, size_t ws_size,
                              hipStream_t stream) {
  const float* x       = (const float*)d_in[0];
  const float* in_w    = (const float*)d_in[1];
  const float* conv_w  = (const float*)d_in[2];
  const float* conv_b  = (const float*)d_in[3];
  const float* xproj_w = (const float*)d_in[4];
  const float* dt_w    = (const float*)d_in[5];
  const float* dt_b    = (const float*)d_in[6];
  const float* alog    = (const float*)d_in[7];
  const float* ds      = (const float*)d_in[8];
  const float* ln_g    = (const float*)d_in[9];
  const float* ln_b    = (const float*)d_in[10];
  const float* out_w   = (const float*)d_in[11];
  float* ws  = (float*)d_ws;
  float* out = (float*)d_out;

  k_inproj <<<dim3(256, 3),   256, 0, stream>>>(x, in_w, ws);
  k_conv   <<<12288,          256, 0, stream>>>(conv_w, conv_b, ws);
  k_xproj  <<<dim3(64, 4, 4), 128, 0, stream>>>(xproj_w, ws);
  k_delta  <<<49152,          256, 0, stream>>>(dt_w, dt_b, ws);
  k_scan   <<<dim3(12, 4, 4), 256, 0, stream>>>(alog, ws);
  k_combine<<<dim3(4096, 4),  192, 0, stream>>>(ds, ln_g, ln_b, ws);
  k_outproj<<<256,            256, 0, stream>>>(out_w, ws, out);
}
